// LocallyConnectedBlock_78709570666928
// MI455X (gfx1250) — matrix operand order repaired and checked by reading, not compiled, not run on hardware
//
#include <hip/hip_runtime.h>
#include <hip/hip_bf16.h>

// Problem constants (match reference)
#define BB   32
#define HH   64
#define WWW  64
#define CIN  32
#define OHH  32
#define OWW  32
#define FF   64
#define KTOT 800          // 5*5*32, k = c*25 + kh*5 + kw (jax patches: channel-major)
#define NPOS 1024         // OH*OW
#define NTOT 32768        // B*OH*OW
#define NSTEP 25          // K steps of 32
#define SA_STRIDE 806     // halfs; 403 banks/row == 19 (mod 64) -> conflict-free frag reads
#define SB_STRIDE 38      // halfs; 19 banks/row -> conflict-free frag reads

typedef __attribute__((ext_vector_type(16))) _Float16 v16h;
typedef __attribute__((ext_vector_type(8)))  float    v8f;

union FragH { v16h v; unsigned u[8]; };
union FragF { v8f  v; float    f[8]; };

static __device__ __forceinline__ unsigned short f2h(float x) {
  return __builtin_bit_cast(unsigned short, (_Float16)x);
}
// one v_cvt_pk_rtz_f16_f32 instead of 6-op cvt/shift/or sequence
static __device__ __forceinline__ unsigned pkh2(float lo, float hi) {
  return __builtin_bit_cast(unsigned, __builtin_amdgcn_cvt_pkrtz(lo, hi));
}

// ---------------------------------------------------------------------------
// Pass 1: per-position GEMM (32x800 @ 800x64) via v_wmma_f32_16x16x32_f16.
// Weight stream (210MB, single use) is NT-loaded into registers, the WMMA for
// the current step issues before the loads are consumed (s_wait_loadcnt lands
// after the wmma), then the data is packed to f16 and ping-pong staged in LDS.
// One barrier per K-step. Fused bias + leaky-ReLU; y -> d_out; deterministic
// per-block channel sum/sumsq partials -> d_ws.
// ---------------------------------------------------------------------------
__global__ __launch_bounds__(256) void lc_wmma_kernel(
    const float* __restrict__ x,      // [32,64,64,32]
    const float* __restrict__ wk,     // [32,32,800,64]
    const float* __restrict__ bias,   // [32,32,64]
    float* __restrict__ out,          // [32,32,32,64]
    float* __restrict__ partials)     // [1024][128]
{
  __shared__ unsigned short sA[32 * SA_STRIDE];      // patches f16 [b][k]
  __shared__ unsigned short sB[2][64 * SB_STRIDE];   // weight chunk f16 [f][kk], ping-pong
  __shared__ float sS[64 * 4];                       // per-channel sum slots
  __shared__ float sQ[64 * 4];                       // per-channel sumsq slots

  const int tid  = threadIdx.x;
  const int lane = tid & 31;
  const int wv   = tid >> 5;
  const int mt   = wv >> 2;        // M tile 0..1 (batch)
  const int nt   = wv & 3;         // N tile 0..3 (channels)
  const int hi   = lane >> 4;      // half-wave
  const int nl   = lane & 15;
  const int n    = nt * 16 + nl;   // output channel 0..63

  const int oh = blockIdx.x >> 5;
  const int ow = blockIdx.x & 31;

  // ---- stage all patches for this position into LDS as f16 (x is L2-resident)
  {
    const int c = tid & 31;                       // lane == channel -> 128B coalesced
    for (int i = tid >> 5; i < 32 * 25; i += 8) { // (b,pos) pairs
      const int b   = i / 25;
      const int pos = i - b * 25;
      const int kh  = pos / 5;
      const int kw  = pos - kh * 5;
      const int h   = (oh << 1) + kh - 1;         // SAME pad: top/left = 1
      const int w   = (ow << 1) + kw - 1;
      float v = 0.0f;
      if ((unsigned)h < 64u && (unsigned)w < 64u)
        v = x[((b * HH + h) * WWW + w) * CIN + c];
      sA[b * SA_STRIDE + c * 25 + pos] = f2h(v);
    }
  }

  FragF acc;
#pragma unroll
  for (int j = 0; j < 8; ++j) acc.f[j] = 0.0f;

  const float* wp = wk + (size_t)blockIdx.x * (KTOT * FF);
  const int f  = tid & 63;            // channel this thread stages
  const int kg = (tid >> 6) << 3;     // k sub-range base: 0,8,16,24

  // prologue: stage step 0 into buffer 0
  {
    const float* wr = wp + (size_t)kg * FF + f;
#pragma unroll
    for (int j = 0; j < 4; ++j) {
      const float w0 = __builtin_nontemporal_load(wr + (2 * j) * FF);
      const float w1 = __builtin_nontemporal_load(wr + (2 * j + 1) * FF);
      *reinterpret_cast<unsigned*>(&sB[0][f * SB_STRIDE + kg + 2 * j]) = pkh2(w0, w1);
    }
  }

  for (int step = 0; step < NSTEP; ++step) {
    const int k0  = step << 5;
    const int cur = step & 1;
    __syncthreads();  // publishes sB[cur] (and sA on step 0); sB[cur^1] free

    // (1) ISSUE next step's NT loads (not consumed yet -> no wait here)
    float wreg[8];
    const bool pf = (step + 1 < NSTEP);
    if (pf) {
      const float* wr = wp + (size_t)(k0 + 32 + kg) * FF + f;
#pragma unroll
      for (int j = 0; j < 8; ++j)
        wreg[j] = __builtin_nontemporal_load(wr + j * FF);
    }

    // (2) build fragments (contiguous k = 16*half + s in A and B) and issue WMMA.
    //     No DS stores precede these loads in program order, so s_wait_dscnt
    //     here covers only the fragment loads, not the global stream.
    FragH a, b;
    const unsigned short* pA = &sA[(mt * 16 + nl) * SA_STRIDE + k0];
    const unsigned short* pB = &sB[cur][n * SB_STRIDE + hi * 16];
#pragma unroll
    for (int v = 0; v < 8; ++v) {
      // A: VGPR v holds K = {2v,2v+1} + 16 for upper half-wave (matches B)
      const int ka = (hi << 4) + (v << 1);
      a.u[v] = *reinterpret_cast<const unsigned*>(pA + ka);
      // B: lanes 0-15 hold K=0..15, lanes 16-31 hold K=16..31; N = lane%16
      b.u[v] = *reinterpret_cast<const unsigned*>(pB + (v << 1));
    }
    acc.v = __builtin_amdgcn_wmma_f32_16x16x32_f16(
        false, a.v, false, b.v, (short)0, acc.v, false, false);

    // (3) consume the prefetched weights: pack f16 pairs, store to other buffer
    if (pf) {
#pragma unroll
      for (int j = 0; j < 4; ++j)
        *reinterpret_cast<unsigned*>(&sB[cur ^ 1][f * SB_STRIDE + kg + 2 * j]) =
            pkh2(wreg[2 * j], wreg[2 * j + 1]);
    }
  }

  // ---- epilogue: bias + leaky relu, store y, per-channel partial stats
  const float bval = bias[blockIdx.x * FF + n];
  float s = 0.0f, q = 0.0f;
#pragma unroll
  for (int j = 0; j < 8; ++j) {
    const int m = mt * 16 + hi * 8 + j;  // batch index (C/D VGPR layout)
    float y = acc.f[j] + bval;
    y = (y >= 0.0f) ? y : 0.01f * y;
    out[((m * OHH + oh) * OWW + ow) * FF + n] = y;
    s += y;
    q += y * y;
  }
  const int slot = n * 4 + mt * 2 + hi;  // unique per thread -> deterministic
  sS[slot] = s;
  sQ[slot] = q;
  __syncthreads();
  if (tid < 128) {
    const int ch = tid & 63;
    const float* base = (tid < 64) ? sS : sQ;
    const float t = base[ch * 4] + base[ch * 4 + 1] +
                    base[ch * 4 + 2] + base[ch * 4 + 3];
    partials[(size_t)blockIdx.x * 128 + tid] = t;
  }
}

// ---------------------------------------------------------------------------
// Pass 2: fold 1024x128 partials -> per-channel BN affine params.
// stats[f] = rsqrt(var+eps)*scale ; stats[64+f] = bn_bias - mean*stats[f]
// ---------------------------------------------------------------------------
__global__ __launch_bounds__(128) void lc_stats_kernel(
    const float* __restrict__ partials, const float* __restrict__ scale,
    const float* __restrict__ bn_bias, float* __restrict__ stats)
{
  const int t = threadIdx.x;  // 128 threads
  float s = 0.0f;
  for (int b = 0; b < NPOS; ++b) s += partials[(size_t)b * 128 + t];
  __shared__ float ls[128];
  ls[t] = s;
  __syncthreads();
  if (t < 64) {
    const float inv_n = 1.0f / (float)NTOT;
    const float mean = ls[t] * inv_n;
    const float var  = ls[64 + t] * inv_n - mean * mean;
    const float a    = rsqrtf(var + 1e-5f) * scale[t];
    stats[t]      = a;
    stats[64 + t] = bn_bias[t] - mean * a;
  }
}

// ---------------------------------------------------------------------------
// Pass 3: in-place normalize, float4 elementwise (memory-trivial: ~17MB).
// ---------------------------------------------------------------------------
__global__ __launch_bounds__(256) void lc_bn_kernel(
    float* __restrict__ out, const float* __restrict__ stats)
{
  __shared__ float a[64], b2[64];
  if (threadIdx.x < 64) a[threadIdx.x] = stats[threadIdx.x];
  else if (threadIdx.x < 128) b2[threadIdx.x - 64] = stats[threadIdx.x];
  __syncthreads();

  const int i = blockIdx.x * 256 + threadIdx.x;   // float4 index
  float4 y = reinterpret_cast<float4*>(out)[i];
  const int f0 = (i << 2) & 63;
  y.x = y.x * a[f0 + 0] + b2[f0 + 0];
  y.y = y.y * a[f0 + 1] + b2[f0 + 1];
  y.z = y.z * a[f0 + 2] + b2[f0 + 2];
  y.w = y.w * a[f0 + 3] + b2[f0 + 3];
  reinterpret_cast<float4*>(out)[i] = y;
}

// ---------------------------------------------------------------------------
extern "C" void kernel_launch(void* const* d_in, const int* in_sizes, int n_in,
                              void* d_out, int out_size, void* d_ws, size_t ws_size,
                              hipStream_t stream) {
  (void)in_sizes; (void)n_in; (void)out_size; (void)ws_size;
  const float* x       = (const float*)d_in[0];  // [32,64,64,32]
  const float* kernel  = (const float*)d_in[1];  // [32,32,800,64]
  const float* bias    = (const float*)d_in[2];  // [32,32,64]
  const float* scale   = (const float*)d_in[3];  // [64]
  const float* bn_bias = (const float*)d_in[4];  // [64]
  float* out = (float*)d_out;

  float* partials = (float*)d_ws;                   // 1024*128 floats (512KB)
  float* stats    = partials + (size_t)NPOS * 128;  // 128 floats

  lc_wmma_kernel<<<NPOS, 256, 0, stream>>>(x, kernel, bias, out, partials);
  lc_stats_kernel<<<1, 128, 0, stream>>>(partials, scale, bn_bias, stats);
  lc_bn_kernel<<<(NTOT * FF) / (256 * 4), 256, 0, stream>>>(out, stats);
}
